// TransformerEncoderLayer_34643206209941
// MI455X (gfx1250) — compile-verified, matrix operand order checked by reading, not run on hardware
//
#include <hip/hip_runtime.h>
#include <cstdint>
#include <cstddef>

// ---------------- problem constants ----------------
#define B_SZ    2
#define L_SEQ   2048
#define DM      1024
#define NH      16
#define DKV     64
#define DFF     4096
#define M_ROWS  (B_SZ * L_SEQ)   // 4096 token rows

// ---------------- WMMA types (gfx1250, wave32) ----------------
typedef __attribute__((ext_vector_type(16))) __bf16 v16bf;
typedef __attribute__((ext_vector_type(8)))  float  v8f;

union Frag16 {            // one 16x32 bf16 A/B fragment slice per lane: 16 bf16 = 32B
  uint4 u[2];
  v16bf v;
};

__device__ __forceinline__ unsigned short f2bf(float f) {
  unsigned int u = __float_as_uint(f);
  u += 0x7FFFu + ((u >> 16) & 1u);      // round-to-nearest-even
  return (unsigned short)(u >> 16);
}

// Per-lane fragment load for 16-bit WMMA operands (ISA 7.12.2 layout):
// lane half (lane>=16) owns K chunks [off8, off8+8) and [16+off8, 16+off8+8)
__device__ __forceinline__ Frag16 load_frag(const unsigned short* rowPtr, int k, int off8) {
  Frag16 f;
  f.u[0] = *reinterpret_cast<const uint4*>(rowPtr + k + off8);
  f.u[1] = *reinterpret_cast<const uint4*>(rowPtr + k + 16 + off8);
  return f;
}

#define WMMA_BF16(A, B, C) \
  __builtin_amdgcn_wmma_f32_16x16x32_bf16(false, (A), false, (B), (short)0, (C), false, false)

// ---------------- fp32 -> bf16 weight conversion ----------------
__global__ void k_cvt_bf16(const float* __restrict__ src, unsigned short* __restrict__ dst, int n) {
  int i = blockIdx.x * blockDim.x + threadIdx.x;
  if (i < n) dst[i] = f2bf(src[i]);
}

// ---------------- RMSNorm (fp32 in, bf16 out) ----------------
__global__ __launch_bounds__(256) void k_rmsnorm(const float* __restrict__ x,
                                                 const float* __restrict__ w,
                                                 unsigned short* __restrict__ yb) {
  const int row = blockIdx.x;
  const int tid = threadIdx.x;
  const float* xr = x + (size_t)row * DM;
  float s = 0.f;
  for (int i = tid; i < DM; i += 256) { float v = xr[i]; s += v * v; }
  __shared__ float red[256];
  red[tid] = s;
  __syncthreads();
  for (int st = 128; st > 0; st >>= 1) {
    if (tid < st) red[tid] += red[tid + st];
    __syncthreads();
  }
  const float scale = rsqrtf(red[0] * (1.0f / DM) + 1e-6f);
  for (int i = tid; i < DM; i += 256)
    yb[(size_t)row * DM + i] = f2bf(xr[i] * scale * w[i]);
}

// ---------------- generic bf16 WMMA GEMM: C[M,N] = A[M,K] * W[N,K]^T ----------------
// block = 256 threads = 8 waves (4m x 2n), block tile 128x128, wave tile 32x64
// per K-step: 12 b128 loads feed 8 WMMAs (A reused 4x, B reused 2x in registers)
__global__ __launch_bounds__(256) void k_gemm(const unsigned short* __restrict__ A,
                                              const unsigned short* __restrict__ W,
                                              int N, int K,
                                              float* __restrict__ outF,
                                              unsigned short* __restrict__ outB,
                                              const float* __restrict__ resid) {
  const int lane = threadIdx.x & 31;
  const int wid  = threadIdx.x >> 5;
  const int m0 = blockIdx.y * 128 + (wid >> 1) * 32;
  const int n0 = blockIdx.x * 128 + (wid & 1)  * 64;
  const int r    = lane & 15;
  const int off8 = (lane >> 4) << 3;

  const unsigned short* a0p = A + (size_t)(m0 + r) * K;
  const unsigned short* a1p = A + (size_t)(m0 + 16 + r) * K;
  const unsigned short* wp[4];
  #pragma unroll
  for (int j = 0; j < 4; ++j) wp[j] = W + (size_t)(n0 + j * 16 + r) * K;

  v8f acc[2][4];
  #pragma unroll
  for (int i = 0; i < 2; ++i)
    #pragma unroll
    for (int j = 0; j < 4; ++j) acc[i][j] = (v8f){};

  for (int k = 0; k < K; k += 32) {
    __builtin_prefetch(a0p + k + 512, 0, 0);   // global_prefetch_b8
    __builtin_prefetch(wp[0] + k + 512, 0, 0);
    Frag16 a0 = load_frag(a0p, k, off8);
    Frag16 a1 = load_frag(a1p, k, off8);
    Frag16 b[4];
    #pragma unroll
    for (int j = 0; j < 4; ++j) b[j] = load_frag(wp[j], k, off8);
    #pragma unroll
    for (int j = 0; j < 4; ++j) {
      acc[0][j] = WMMA_BF16(a0.v, b[j].v, acc[0][j]);
      acc[1][j] = WMMA_BF16(a1.v, b[j].v, acc[1][j]);
    }
  }

  // C/D layout: VGPR e -> row e (+8 for lanes>=16), col = lane&15
  #pragma unroll
  for (int i = 0; i < 2; ++i) {
    #pragma unroll
    for (int j = 0; j < 4; ++j) {
      #pragma unroll
      for (int e = 0; e < 8; ++e) {
        const int row = m0 + i * 16 + e + off8;
        const int col = n0 + j * 16 + r;
        float v = acc[i][j][e];
        if (resid) v += resid[(size_t)row * N + col];
        if (outF)  outF[(size_t)row * N + col] = v;
        if (outB)  outB[(size_t)row * N + col] = f2bf(v);
      }
    }
  }
}

// ---------------- fused gated MLP GEMM: g = gelu(A*W0^T) * (A*W1^T), bf16 out ----------------
// same 32x64 wave tile per weight stream: 20 b128 loads feed 16 WMMAs per K-step
__global__ __launch_bounds__(256) void k_gemm_gated(const unsigned short* __restrict__ A,
                                                    const unsigned short* __restrict__ W0,
                                                    const unsigned short* __restrict__ W1,
                                                    int N, int K,
                                                    unsigned short* __restrict__ outB) {
  const int lane = threadIdx.x & 31;
  const int wid  = threadIdx.x >> 5;
  const int m0 = blockIdx.y * 128 + (wid >> 1) * 32;
  const int n0 = blockIdx.x * 128 + (wid & 1)  * 64;
  const int r    = lane & 15;
  const int off8 = (lane >> 4) << 3;

  const unsigned short* a0p = A + (size_t)(m0 + r) * K;
  const unsigned short* a1p = A + (size_t)(m0 + 16 + r) * K;
  const unsigned short* up[4];
  const unsigned short* gp[4];
  #pragma unroll
  for (int j = 0; j < 4; ++j) {
    up[j] = W0 + (size_t)(n0 + j * 16 + r) * K;
    gp[j] = W1 + (size_t)(n0 + j * 16 + r) * K;
  }

  v8f hAcc[2][4], gAcc[2][4];
  #pragma unroll
  for (int i = 0; i < 2; ++i)
    #pragma unroll
    for (int j = 0; j < 4; ++j) { hAcc[i][j] = (v8f){}; gAcc[i][j] = (v8f){}; }

  for (int k = 0; k < K; k += 32) {
    Frag16 a0 = load_frag(a0p, k, off8);
    Frag16 a1 = load_frag(a1p, k, off8);
    #pragma unroll
    for (int j = 0; j < 4; ++j) {
      Frag16 bu = load_frag(up[j], k, off8);
      hAcc[0][j] = WMMA_BF16(a0.v, bu.v, hAcc[0][j]);
      hAcc[1][j] = WMMA_BF16(a1.v, bu.v, hAcc[1][j]);
      Frag16 bg = load_frag(gp[j], k, off8);
      gAcc[0][j] = WMMA_BF16(a0.v, bg.v, gAcc[0][j]);
      gAcc[1][j] = WMMA_BF16(a1.v, bg.v, gAcc[1][j]);
    }
  }

  #pragma unroll
  for (int i = 0; i < 2; ++i) {
    #pragma unroll
    for (int j = 0; j < 4; ++j) {
      #pragma unroll
      for (int e = 0; e < 8; ++e) {
        const int row = m0 + i * 16 + e + off8;
        const int col = n0 + j * 16 + r;
        const float h = hAcc[i][j][e];
        const float act = 0.5f * h * (1.0f + erff(h * 0.70710678118654752f));  // exact gelu
        outB[(size_t)row * N + col] = f2bf(act * gAcc[i][j][e]);
      }
    }
  }
}

// ---------------- V transpose: [B,L,H,64] -> [B,H,64,L] (bf16) ----------------
__global__ void k_transpose_v(const unsigned short* __restrict__ v,
                              unsigned short* __restrict__ vt) {
  const unsigned int i = blockIdx.x * blockDim.x + threadIdx.x;
  if (i >= (unsigned)(B_SZ * NH * DKV * L_SEQ)) return;
  const int l = i & (L_SEQ - 1);
  const int d = (i >> 11) & 63;
  const int h = (i >> 17) & 15;
  const int b = (int)(i >> 21);
  vt[i] = v[((size_t)(b * L_SEQ + l)) * DM + h * DKV + d];
}

// ---------------- flash-style attention, one wave per (b, h, 16-query tile) ----------------
#define CHUNK 256
__global__ __launch_bounds__(32) void k_attention(const unsigned short* __restrict__ Q,
                                                  const unsigned short* __restrict__ Kmat,
                                                  const unsigned short* __restrict__ VT,
                                                  const float* __restrict__ mask,
                                                  unsigned short* __restrict__ Obf) {
  const int qt = blockIdx.x;      // L/16 query tiles
  const int h  = blockIdx.y;
  const int b  = blockIdx.z;
  const int lane = threadIdx.x;
  const int r    = lane & 15;
  const int off8 = (lane >> 4) << 3;
  const int qbase = qt * 16;

  __shared__ __align__(16) float          S[16][CHUNK];
  __shared__ __align__(16) unsigned short P[16][CHUNK];
  __shared__ float corr[16];
  __shared__ float invs[16];

  // Q fragments (d=64 -> two k-steps of 32), lane r = query row
  const unsigned short* qrow = Q + ((size_t)(b * L_SEQ + qbase + r)) * DM + h * DKV;
  Frag16 aq0 = load_frag(qrow, 0, off8);
  Frag16 aq1 = load_frag(qrow, 32, off8);

  v8f o0 = {}, o1 = {}, o2 = {}, o3 = {};
  float runMax = -3.4e38f, runSum = 0.f;    // meaningful in lanes 0..15 only

  for (int c = 0; c < L_SEQ; c += CHUNK) {
    // ---- S = Q * K^T for this key chunk, via WMMA, tile by 16 keys ----
    for (int kt = 0; kt < CHUNK; kt += 16) {
      const unsigned short* krow = Kmat + ((size_t)(b * L_SEQ + c + kt + r)) * DM + h * DKV;
      Frag16 bk0 = load_frag(krow, 0, off8);
      Frag16 bk1 = load_frag(krow, 32, off8);
      v8f s = {};
      s = WMMA_BF16(aq0.v, bk0.v, s);
      s = WMMA_BF16(aq1.v, bk1.v, s);
      #pragma unroll
      for (int e = 0; e < 8; ++e)
        S[e + off8][kt + r] = s[e];
    }
    __syncthreads();

    // ---- online softmax update: lane t (< 16) owns query row t ----
    if (lane < 16) {
      const float* mrow = mask + (size_t)(qbase + lane) * L_SEQ + c;
      float mx = -3.4e38f;
      for (int j = 0; j < CHUNK; ++j) {
        float t = S[lane][j] + mrow[j];
        S[lane][j] = t;
        mx = fmaxf(mx, t);
      }
      float mNew = fmaxf(runMax, mx);
      float cr = __expf(runMax - mNew);
      float sum = 0.f;
      for (int j = 0; j < CHUNK; ++j) {
        float pv = __expf(S[lane][j] - mNew);
        P[lane][j] = f2bf(pv);
        sum += pv;
      }
      runSum = runSum * cr + sum;
      runMax = mNew;
      corr[lane] = cr;
    }
    __syncthreads();

    // ---- rescale existing O accumulators ----
    #pragma unroll
    for (int e = 0; e < 8; ++e) {
      const float f = corr[e + off8];
      o0[e] *= f; o1[e] *= f; o2[e] *= f; o3[e] *= f;
    }

    // ---- O += P * V over this chunk (K = 32 keys per WMMA) ----
    for (int ks = 0; ks < CHUNK; ks += 32) {
      Frag16 pa;
      pa.u[0] = *reinterpret_cast<const uint4*>(&P[r][ks + off8]);
      pa.u[1] = *reinterpret_cast<const uint4*>(&P[r][ks + 16 + off8]);
      #pragma unroll
      for (int dt = 0; dt < 4; ++dt) {
        const unsigned short* vrow =
            VT + ((size_t)((b * NH + h) * DKV + dt * 16 + r)) * L_SEQ + c + ks;
        Frag16 bv;
        bv.u[0] = *reinterpret_cast<const uint4*>(vrow + off8);
        bv.u[1] = *reinterpret_cast<const uint4*>(vrow + 16 + off8);
        switch (dt) {
          case 0: o0 = WMMA_BF16(pa.v, bv.v, o0); break;
          case 1: o1 = WMMA_BF16(pa.v, bv.v, o1); break;
          case 2: o2 = WMMA_BF16(pa.v, bv.v, o2); break;
          default: o3 = WMMA_BF16(pa.v, bv.v, o3); break;
        }
      }
    }
    __syncthreads();
  }

  if (lane < 16) invs[lane] = 1.0f / runSum;
  __syncthreads();

  #pragma unroll
  for (int e = 0; e < 8; ++e) {
    const float f = invs[e + off8];
    const size_t base = ((size_t)(b * L_SEQ + qbase + e + off8)) * DM + h * DKV + r;
    Obf[base +  0] = f2bf(o0[e] * f);
    Obf[base + 16] = f2bf(o1[e] * f);
    Obf[base + 32] = f2bf(o2[e] * f);
    Obf[base + 48] = f2bf(o3[e] * f);
  }
}

// ---------------- host-side orchestration ----------------
extern "C" void kernel_launch(void* const* d_in, const int* in_sizes, int n_in,
                              void* d_out, int out_size, void* d_ws, size_t ws_size,
                              hipStream_t stream) {
  (void)in_sizes; (void)n_in; (void)out_size; (void)ws_size;
  const float* x    = (const float*)d_in[0];
  const float* mask = (const float*)d_in[1];
  const float* wq   = (const float*)d_in[2];
  const float* wk   = (const float*)d_in[3];
  const float* wv   = (const float*)d_in[4];
  const float* wo   = (const float*)d_in[5];
  const float* ln1  = (const float*)d_in[6];
  const float* ln2  = (const float*)d_in[7];
  const float* wi0  = (const float*)d_in[8];
  const float* wi1  = (const float*)d_in[9];
  const float* wom  = (const float*)d_in[10];

  char* base = (char*)d_ws;
  size_t off = 0;
  auto carveB = [&](size_t elems) -> unsigned short* {   // bf16 region
    unsigned short* q = (unsigned short*)(base + off);
    off = (off + elems * 2 + 255) & ~(size_t)255;
    return q;
  };
  auto carveF = [&](size_t elems) -> float* {            // fp32 region
    float* q = (float*)(base + off);
    off = (off + elems * 4 + 255) & ~(size_t)255;
    return q;
  };

  unsigned short* wq_b  = carveB((size_t)DM * DM);
  unsigned short* wk_b  = carveB((size_t)DM * DM);
  unsigned short* wv_b  = carveB((size_t)DM * DM);
  unsigned short* wo_b  = carveB((size_t)DM * DM);
  unsigned short* wi0_b = carveB((size_t)DFF * DM);
  unsigned short* wi1_b = carveB((size_t)DFF * DM);
  unsigned short* wom_b = carveB((size_t)DM * DFF);
  unsigned short* y1_b  = carveB((size_t)M_ROWS * DM);
  unsigned short* q_b   = carveB((size_t)M_ROWS * DM);
  unsigned short* k_b   = carveB((size_t)M_ROWS * DM);
  unsigned short* v_b   = carveB((size_t)M_ROWS * DM);
  unsigned short* vt_b  = carveB((size_t)M_ROWS * DM);
  unsigned short* at_b  = carveB((size_t)M_ROWS * DM);
  unsigned short* y2_b  = carveB((size_t)M_ROWS * DM);
  unsigned short* g_b   = carveB((size_t)M_ROWS * DFF);
  float*          x2    = carveF((size_t)M_ROWS * DM);

  // 1) weights fp32 -> bf16
  auto cvt = [&](const float* s, unsigned short* d, int n) {
    k_cvt_bf16<<<dim3((n + 255) / 256), dim3(256), 0, stream>>>(s, d, n);
  };
  cvt(wq,  wq_b,  DM * DM);
  cvt(wk,  wk_b,  DM * DM);
  cvt(wv,  wv_b,  DM * DM);
  cvt(wo,  wo_b,  DM * DM);
  cvt(wi0, wi0_b, DFF * DM);
  cvt(wi1, wi1_b, DFF * DM);
  cvt(wom, wom_b, DM * DFF);

  // 2) y1 = rmsnorm(x, ln1)  (bf16)
  k_rmsnorm<<<dim3(M_ROWS), dim3(256), 0, stream>>>(x, ln1, y1_b);

  // 3) Q/K/V projections (bf16 WMMA GEMMs), N = 1024, K = 1024
  dim3 gProj(DM / 128, M_ROWS / 128);
  k_gemm<<<gProj, dim3(256), 0, stream>>>(y1_b, wq_b, DM, DM, nullptr, q_b, nullptr);
  k_gemm<<<gProj, dim3(256), 0, stream>>>(y1_b, wk_b, DM, DM, nullptr, k_b, nullptr);
  k_gemm<<<gProj, dim3(256), 0, stream>>>(y1_b, wv_b, DM, DM, nullptr, v_b, nullptr);

  // 4) V transpose for P*V fragment loads
  k_transpose_v<<<dim3((B_SZ * NH * DKV * L_SEQ + 255) / 256), dim3(256), 0, stream>>>(v_b, vt_b);

  // 5) attention (flash-style, WMMA for both matmuls)
  k_attention<<<dim3(L_SEQ / 16, NH, B_SZ), dim3(32), 0, stream>>>(q_b, k_b, vt_b, mask, at_b);

  // 6) x2 = x + attn @ wo^T  (fp32 out)
  k_gemm<<<gProj, dim3(256), 0, stream>>>(at_b, wo_b, DM, DM, x2, nullptr, x);

  // 7) y2 = rmsnorm(x2, ln2)  (bf16)
  k_rmsnorm<<<dim3(M_ROWS), dim3(256), 0, stream>>>(x2, ln2, y2_b);

  // 8) gated MLP: g = gelu(y2@wi0^T) * (y2@wi1^T)  (bf16), N = 4096
  k_gemm_gated<<<dim3(DFF / 128, M_ROWS / 128), dim3(256), 0, stream>>>(y2_b, wi0_b, wi1_b, DFF, DM, g_b);

  // 9) out = x2 + g @ wo_mlp^T  (fp32 -> d_out), K = 4096
  k_gemm<<<gProj, dim3(256), 0, stream>>>(g_b, wom_b, DM, DFF, (float*)d_out, nullptr, x2);
}
